// TRANSFORMER_CRF_83983790506057
// MI455X (gfx1250) — matrix operand order repaired and checked by reading, not compiled, not run on hardware
//
#include <hip/hip_runtime.h>
#include <hip/hip_bf16.h>
#include <stdint.h>

typedef __attribute__((ext_vector_type(8)))  float        v8f;
typedef __attribute__((ext_vector_type(8)))  __bf16       v8bf;
typedef __attribute__((ext_vector_type(16))) __bf16       v16bf;
typedef __attribute__((ext_vector_type(4)))  unsigned int v4u;
typedef __attribute__((ext_vector_type(4)))  int          v4i;
typedef __attribute__((ext_vector_type(8)))  int          v8i;

#define S_LEN 2048
#define D_DIM 512
#define H_NUM 8
#define DH    64
#define FF_DIM 2048
#define L_TAG 32
#define NEGV  -10000.0f

#if __has_builtin(__builtin_amdgcn_tensor_load_to_lds) && \
    __has_builtin(__builtin_amdgcn_s_wait_tensorcnt)
#define HAVE_TDM 1
#else
#define HAVE_TDM 0
#endif

// ---------------------------------------------------------------------------
// LDS byte offset of a shared-memory pointer (generic -> addrspace(3))
// ---------------------------------------------------------------------------
__device__ __forceinline__ unsigned lds_offset_of(const void* p)
{
    return (unsigned)(unsigned long long)
        (const __attribute__((address_space(3))) void*)p;
}

#if HAVE_TDM
// ---------------------------------------------------------------------------
// TDM 2D tile load: bf16 tensor [tensor_h, tensor_w] (row stride = tensor_w),
// tile [tile_h, tile_w=32] starting at gaddr -> LDS at ldsoff, with LDS row
// padding 64B+16B (pad_interval code 3 = 16 DWORDs, pad_amount code 3 = 4
// DWORDs) to match the __bf16 [rows][40] LDS layout. OOB rows read as zero.
// ---------------------------------------------------------------------------
__device__ __forceinline__ void tdm_load_2d(unsigned long long gaddr,
                                            unsigned ldsoff,
                                            int tensor_w, int tensor_h,
                                            int tile_h)
{
    v4u g0;
    g0[0] = 1u;                                  // count=1, user descriptor
    g0[1] = ldsoff;                              // lds_addr
    g0[2] = (unsigned)gaddr;                     // global_addr[31:0]
    g0[3] = ((unsigned)(gaddr >> 32) & 0x1FFFFFFu) | (2u << 30); // addr hi, type=2

    int w0 = (1 << 16)    // data_size = 1 (2 bytes)
           | (1 << 20)    // pad_enable
           | (3 << 22)    // pad_interval: 16 DWORDs (64B rows)
           | (3 << 25);   // pad_amount: 4 DWORDs (16B pad)
    v8i g1;
    g1[0] = w0;
    g1[1] = (tensor_w & 0xFFFF) << 16;                           // td0 lo16
    g1[2] = ((tensor_w >> 16) & 0xFFFF) | ((tensor_h & 0xFFFF) << 16);
    g1[3] = ((tensor_h >> 16) & 0xFFFF) | (32 << 16);            // tile_dim0=32
    g1[4] = tile_h;                                              // tile_dim1
    g1[5] = tensor_w;                                            // stride0 lo32
    g1[6] = 0;
    g1[7] = 0;

    v4i z4 = {};
#if defined(__clang_major__) && (__clang_major__ >= 23)
    v8i z8 = {};
    __builtin_amdgcn_tensor_load_to_lds(g0, g1, z4, z4, z8, 0);
#else
    __builtin_amdgcn_tensor_load_to_lds(g0, g1, z4, z4, 0);
#endif
}
#endif

// ---------------------------------------------------------------------------
// f32 -> bf16 elementwise (weight pre-conversion)
// ---------------------------------------------------------------------------
__global__ __launch_bounds__(256) void cvt_bf16_k(
    const float* __restrict__ in, __bf16* __restrict__ out, int n)
{
    int i = blockIdx.x * 256 + threadIdx.x;
    if (i < n) out[i] = (__bf16)in[i];
}

// ---------------------------------------------------------------------------
// Embedding gather: x[s][d] = embed[sentence[s]][d] (f32 + bf16 mirror)
// ---------------------------------------------------------------------------
__global__ __launch_bounds__(256) void embed_gather_k(
    const int* __restrict__ sent, const float* __restrict__ embed,
    float* __restrict__ x, __bf16* __restrict__ xb)
{
    int s = blockIdx.x;
    int tid = threadIdx.x;
    int w = sent[s];
#pragma unroll
    for (int i = 0; i < 2; ++i) {
        int d = tid + i * 256;
        float v = embed[(size_t)w * D_DIM + d];
        x[(size_t)s * D_DIM + d] = v;
        xb[(size_t)s * D_DIM + d] = (__bf16)v;
    }
}

// ---------------------------------------------------------------------------
// WMMA GEMM: C[M,N] = A[M,K] @ W[N,K]^T + bias. A, W are bf16.
// Block tile 128(M) x 64(N), K-step 32, 8 waves. Tiles staged to LDS by the
// Tensor Data Mover (wave 0 issues descriptors, waits TENSORcnt, barrier).
// ---------------------------------------------------------------------------
template <bool RELU, bool WF32, bool WBF16>
__global__ __launch_bounds__(256) void gemm_wmma_k(
    const __bf16* __restrict__ A, const __bf16* __restrict__ W,
    const float* __restrict__ bias,
    float* __restrict__ Cf, __bf16* __restrict__ Cb,
    int M, int N, int K)
{
    __shared__ __bf16 As[128][40];
    __shared__ __bf16 Bs[64][40];

    int m0 = blockIdx.y * 128;
    int n0 = blockIdx.x * 64;
    int tid = threadIdx.x;
    int wave = tid >> 5, lane = tid & 31;
    int wm = (wave & 3) * 32, wn = (wave >> 2) * 32;
    int lg = lane >> 4, lm = lane & 15;

    v8f acc[2][2] = {};

    for (int k0 = 0; k0 < K; k0 += 32) {
#if HAVE_TDM
        if (wave == 0) {
            unsigned long long ga = (unsigned long long)(uintptr_t)A
                                  + ((unsigned long long)m0 * K + k0) * 2ull;
            tdm_load_2d(ga, lds_offset_of(&As[0][0]), K, M, 128);
            unsigned long long gw = (unsigned long long)(uintptr_t)W
                                  + ((unsigned long long)n0 * K + k0) * 2ull;
            tdm_load_2d(gw, lds_offset_of(&Bs[0][0]), K, N, 64);
            __builtin_amdgcn_s_wait_tensorcnt(0);
        }
#else
        {
            int row = tid >> 1, colb = (tid & 1) * 16;
            const __bf16* Ap = A + (size_t)(m0 + row) * K + k0 + colb;
            *(v8bf*)&As[row][colb]     = *(const v8bf*)Ap;
            *(v8bf*)&As[row][colb + 8] = *(const v8bf*)(Ap + 8);
        }
        if (tid < 128) {
            int row = tid >> 1, colb = (tid & 1) * 16;
            int n = n0 + row;
            if (n < N) {
                const __bf16* Wp = W + (size_t)n * K + k0 + colb;
                *(v8bf*)&Bs[row][colb]     = *(const v8bf*)Wp;
                *(v8bf*)&Bs[row][colb + 8] = *(const v8bf*)(Wp + 8);
            } else {
                v8bf z = {};
                *(v8bf*)&Bs[row][colb]     = z;
                *(v8bf*)&Bs[row][colb + 8] = z;
            }
        }
#endif
        __syncthreads();

        v16bf a[2], b[2];
#pragma unroll
        for (int i = 0; i < 2; ++i) {
            int r = wm + i * 16 + lm;
            *(v8bf*)&a[i]       = *(const v8bf*)&As[r][8 * lg];
            *((v8bf*)&a[i] + 1) = *(const v8bf*)&As[r][16 + 8 * lg];
        }
#pragma unroll
        for (int j = 0; j < 2; ++j) {
            int c = wn + j * 16 + lm;
            *(v8bf*)&b[j]       = *(const v8bf*)&Bs[c][8 * lg];
            *((v8bf*)&b[j] + 1) = *(const v8bf*)&Bs[c][16 + 8 * lg];
        }
#pragma unroll
        for (int i = 0; i < 2; ++i)
#pragma unroll
            for (int j = 0; j < 2; ++j)
                acc[i][j] = __builtin_amdgcn_wmma_f32_16x16x32_bf16(
                    false, a[i], false, b[j], (short)0, acc[i][j], false, false);
        __syncthreads();
    }

#pragma unroll
    for (int i = 0; i < 2; ++i) {
#pragma unroll
        for (int j = 0; j < 2; ++j) {
            int n = n0 + wn + j * 16 + lm;
            if (n >= N) continue;
            float bv = bias ? bias[n] : 0.0f;
#pragma unroll
            for (int r = 0; r < 8; ++r) {
                int m = m0 + wm + i * 16 + r + 8 * lg;
                float v = acc[i][j][r] + bv;
                if (RELU) v = fmaxf(v, 0.0f);
                if (WF32)  Cf[(size_t)m * N + n] = v;
                if (WBF16) Cb[(size_t)m * N + n] = (__bf16)v;
            }
        }
    }
}

// ---------------------------------------------------------------------------
// Transpose V into vt[h][dh][s] (bf16) for contiguous flash B-fragments
// ---------------------------------------------------------------------------
__global__ __launch_bounds__(256) void transpose_v_k(
    const __bf16* __restrict__ qkv, __bf16* __restrict__ vt)
{
    int idx = blockIdx.x * 256 + threadIdx.x;   // [h][s][d]
    int d = idx & 63;
    int s = (idx >> 6) & 2047;
    int h = idx >> 17;
    vt[((size_t)h * DH + d) * S_LEN + s] =
        qkv[(size_t)s * (3 * D_DIM) + 2 * D_DIM + h * DH + d];
}

// ---------------------------------------------------------------------------
// Flash attention: one wave per (16 query rows, head). Online softmax.
// Output written directly as bf16 (feeds the Wo projection GEMM).
// ---------------------------------------------------------------------------
__global__ __launch_bounds__(32) void flash_attn_k(
    const __bf16* __restrict__ qkv, const __bf16* __restrict__ vt,
    __bf16* __restrict__ ob)
{
    int h = blockIdx.y;
    int m0 = blockIdx.x * 16;
    int lane = threadIdx.x;
    int lg = lane >> 4, lm = lane & 15;

    __shared__ __bf16 Pt[16][40];

    v16bf aq[2];
    {
        const __bf16* qp = qkv + (size_t)(m0 + lm) * (3 * D_DIM) + h * DH;
#pragma unroll
        for (int ks = 0; ks < 2; ++ks) {
            *(v8bf*)&aq[ks]       = *(const v8bf*)(qp + ks * 32 + 8 * lg);
            *((v8bf*)&aq[ks] + 1) = *(const v8bf*)(qp + ks * 32 + 16 + 8 * lg);
        }
    }

    float mi[8], li[8];
#pragma unroll
    for (int r = 0; r < 8; ++r) { mi[r] = -1e30f; li[r] = 0.0f; }
    v8f acc[4] = {};

    for (int j0 = 0; j0 < S_LEN; j0 += 32) {
        if (j0 + 32 < S_LEN) {  // prefetch next K/V block
            __builtin_prefetch(qkv + (size_t)(j0 + 32 + lm) * (3 * D_DIM)
                               + D_DIM + h * DH, 0, 1);
            __builtin_prefetch(vt + ((size_t)h * DH + lm) * S_LEN + j0 + 32, 0, 1);
        }
        // ---- scores: two 16x16 tiles ----
        v8f s[2];
#pragma unroll
        for (int t = 0; t < 2; ++t) {
            const __bf16* kp = qkv + (size_t)(j0 + t * 16 + lm) * (3 * D_DIM)
                               + D_DIM + h * DH;
            v16bf bk0, bk1;
            *(v8bf*)&bk0       = *(const v8bf*)(kp + 8 * lg);
            *((v8bf*)&bk0 + 1) = *(const v8bf*)(kp + 16 + 8 * lg);
            *(v8bf*)&bk1       = *(const v8bf*)(kp + 32 + 8 * lg);
            *((v8bf*)&bk1 + 1) = *(const v8bf*)(kp + 32 + 16 + 8 * lg);
            v8f sv = {};
            sv = __builtin_amdgcn_wmma_f32_16x16x32_bf16(
                false, aq[0], false, bk0, (short)0, sv, false, false);
            sv = __builtin_amdgcn_wmma_f32_16x16x32_bf16(
                false, aq[1], false, bk1, (short)0, sv, false, false);
            s[t] = sv;
        }
        // ---- online softmax (rows r+8*lg, 16 lanes per row) ----
#pragma unroll
        for (int r = 0; r < 8; ++r) {
            float sc0 = s[0][r] * 0.125f, sc1 = s[1][r] * 0.125f;
            float mx = fmaxf(sc0, sc1);
#pragma unroll
            for (int d = 1; d < 16; d <<= 1) mx = fmaxf(mx, __shfl_xor(mx, d, 16));
            float nm = fmaxf(mi[r], mx);
            float scale = __expf(mi[r] - nm);
            float p0 = __expf(sc0 - nm), p1 = __expf(sc1 - nm);
            float rs = p0 + p1;
#pragma unroll
            for (int d = 1; d < 16; d <<= 1) rs += __shfl_xor(rs, d, 16);
            li[r] = li[r] * scale + rs;
            mi[r] = nm;
#pragma unroll
            for (int d4 = 0; d4 < 4; ++d4) acc[d4][r] *= scale;
            Pt[r + 8 * lg][lm]      = (__bf16)p0;
            Pt[r + 8 * lg][lm + 16] = (__bf16)p1;
        }
        __syncthreads();
        // ---- P@V: P A-frag via LDS relayout, V B-frags from vt ----
        v16bf ap;
        *(v8bf*)&ap       = *(const v8bf*)&Pt[lm][8 * lg];
        *((v8bf*)&ap + 1) = *(const v8bf*)&Pt[lm][16 + 8 * lg];
#pragma unroll
        for (int d4 = 0; d4 < 4; ++d4) {
            const __bf16* vp = vt + ((size_t)h * DH + d4 * 16 + lm) * S_LEN
                               + j0 + 8 * lg;
            v16bf bv;
            *(v8bf*)&bv       = *(const v8bf*)vp;
            *((v8bf*)&bv + 1) = *(const v8bf*)(vp + 16);
            acc[d4] = __builtin_amdgcn_wmma_f32_16x16x32_bf16(
                false, ap, false, bv, (short)0, acc[d4], false, false);
        }
        __syncthreads();
    }

#pragma unroll
    for (int d4 = 0; d4 < 4; ++d4)
#pragma unroll
        for (int r = 0; r < 8; ++r) {
            int m = m0 + r + 8 * lg;
            ob[(size_t)m * D_DIM + h * DH + d4 * 16 + lm] =
                (__bf16)(acc[d4][r] / li[r]);
        }
}

// ---------------------------------------------------------------------------
// Residual + LayerNorm over D=512 per row: out = LN(a+b)*g + be
// ---------------------------------------------------------------------------
__global__ __launch_bounds__(256) void ln_residual_k(
    const float* __restrict__ a, const float* __restrict__ b,
    const float* __restrict__ g, const float* __restrict__ be,
    float* __restrict__ outf, __bf16* __restrict__ outb)
{
    int row = blockIdx.x;
    int tid = threadIdx.x;
    __shared__ float red[2][8];

    float x0 = a[(size_t)row * D_DIM + tid] + b[(size_t)row * D_DIM + tid];
    float x1 = a[(size_t)row * D_DIM + 256 + tid] + b[(size_t)row * D_DIM + 256 + tid];
    float sum = x0 + x1, sq = x0 * x0 + x1 * x1;
#pragma unroll
    for (int d = 1; d < 32; d <<= 1) {
        sum += __shfl_xor(sum, d, 32);
        sq  += __shfl_xor(sq,  d, 32);
    }
    int wave = tid >> 5, lane = tid & 31;
    if (lane == 0) { red[0][wave] = sum; red[1][wave] = sq; }
    __syncthreads();
    if (tid == 0) {
        float s0 = 0.0f, q0 = 0.0f;
#pragma unroll
        for (int i = 0; i < 8; ++i) { s0 += red[0][i]; q0 += red[1][i]; }
        red[0][0] = s0; red[1][0] = q0;
    }
    __syncthreads();
    float mean = red[0][0] * (1.0f / D_DIM);
    float var  = red[1][0] * (1.0f / D_DIM) - mean * mean;
    float inv  = rsqrtf(var + 1e-5f);
    float y0 = (x0 - mean) * inv * g[tid]       + be[tid];
    float y1 = (x1 - mean) * inv * g[256 + tid] + be[256 + tid];
    if (outf) {
        outf[(size_t)row * D_DIM + tid]       = y0;
        outf[(size_t)row * D_DIM + 256 + tid] = y1;
    }
    if (outb) {
        outb[(size_t)row * D_DIM + tid]       = (__bf16)y0;
        outb[(size_t)row * D_DIM + 256 + tid] = (__bf16)y1;
    }
}

// ---------------------------------------------------------------------------
// Viterbi: one wave, lane = tag. Strict > keeps jnp.argmax first-max rule.
// ---------------------------------------------------------------------------
__global__ __launch_bounds__(32) void viterbi_k(
    const float* __restrict__ feats, const float* __restrict__ T,
    int* __restrict__ bps, float* __restrict__ out)
{
    int lane = threadIdx.x;
    __shared__ float sprev[32];
    __shared__ float term[32];

    float trow[32];
#pragma unroll
    for (int p = 0; p < 32; ++p) trow[p] = T[lane * 32 + p];

    sprev[lane] = (lane == (L_TAG - 2)) ? 0.0f : NEGV;   // START = 30
    __syncthreads();

    for (int t = 0; t < S_LEN; ++t) {
        float best = -3.4e38f; int bi = 0;
#pragma unroll
        for (int p = 0; p < 32; ++p) {
            float v = sprev[p] + trow[p];
            if (v > best) { best = v; bi = p; }
        }
        float nv = best + feats[(size_t)t * 32 + lane];
        bps[(size_t)t * 32 + lane] = bi;
        __syncthreads();
        sprev[lane] = nv;
        __syncthreads();
    }

    term[lane] = sprev[lane] + T[(L_TAG - 1) * 32 + lane];  // STOP = 31
    __syncthreads();
    __threadfence_block();

    if (lane == 0) {
        float best = term[0]; int bt = 0;
        for (int n = 1; n < 32; ++n)
            if (term[n] > best) { best = term[n]; bt = n; }
        out[0] = best;
        int tag = bt;
        out[1 + (S_LEN - 1)] = (float)tag;
        for (int t = S_LEN - 2; t >= 0; --t) {
            tag = bps[(size_t)(t + 1) * 32 + tag];
            out[1 + t] = (float)tag;
        }
    }
}

// ---------------------------------------------------------------------------
// Orchestration
// ---------------------------------------------------------------------------
extern "C" void kernel_launch(void* const* d_in, const int* in_sizes, int n_in,
                              void* d_out, int out_size, void* d_ws, size_t ws_size,
                              hipStream_t stream)
{
    const int*   sent  = (const int*)  d_in[0];
    const float* embed = (const float*)d_in[1];
    const float* Wqkv  = (const float*)d_in[2];
    const float* bqkv  = (const float*)d_in[3];
    const float* Wo    = (const float*)d_in[4];
    const float* bo    = (const float*)d_in[5];
    const float* ln1g  = (const float*)d_in[6];
    const float* ln1b  = (const float*)d_in[7];
    const float* W1    = (const float*)d_in[8];
    const float* b1    = (const float*)d_in[9];
    const float* W2    = (const float*)d_in[10];
    const float* b2    = (const float*)d_in[11];
    const float* ln2g  = (const float*)d_in[12];
    const float* ln2b  = (const float*)d_in[13];
    const float* Wt    = (const float*)d_in[14];
    const float* bt    = (const float*)d_in[15];
    const float* trans = (const float*)d_in[16];
    float* out = (float*)d_out;

    const size_t MB = 1024u * 1024u;
    const size_t KB = 1024u;
    char* ws = (char*)d_ws;
    float*  x     = (float*) (ws + 0);                    // 4 MiB
    __bf16* xb    = (__bf16*)(ws + 4 * MB);               // 2 MiB
    __bf16* qkvb  = (__bf16*)(ws + 6 * MB);               // 6 MiB
    __bf16* vt    = (__bf16*)(ws + 12 * MB);              // 2 MiB
    __bf16* ob    = (__bf16*)(ws + 14 * MB);              // 2 MiB
    float*  po    = (float*) (ws + 16 * MB);              // 4 MiB
    float*  h1    = (float*) (ws + 20 * MB);              // 4 MiB
    __bf16* h1b   = (__bf16*)(ws + 24 * MB);              // 2 MiB
    __bf16* wqkvb = (__bf16*)(ws + 26 * MB);              // 1.5 MiB
    __bf16* wob   = (__bf16*)(ws + 27 * MB + 512 * KB);   // 0.5 MiB
    __bf16* w1b   = (__bf16*)(ws + 28 * MB);              // 2 MiB
    __bf16* w2b   = (__bf16*)(ws + 30 * MB);              // 2 MiB
    __bf16* wtb   = (__bf16*)(ws + 32 * MB);              // 32 KiB
    float*  feats = (float*) (ws + 32 * MB + 256 * KB);   // 256 KiB
    int*    bps   = (int*)   (ws + 32 * MB + 512 * KB);   // 256 KiB
    // reuse of [0, 14 MiB) after attention + LN1:
    __bf16* a1    = (__bf16*)(ws + 0);                    // 8 MiB
    float*  f2    = (float*) (ws + 8 * MB);               // 4 MiB
    __bf16* h2b   = (__bf16*)(ws + 12 * MB);              // 2 MiB

    // 0. weight pre-conversion to bf16 (enables TDM byte-copy staging)
    cvt_bf16_k<<<3072, 256, 0, stream>>>(Wqkv, wqkvb, 1536 * 512);
    cvt_bf16_k<<<1024, 256, 0, stream>>>(Wo,   wob,   512 * 512);
    cvt_bf16_k<<<4096, 256, 0, stream>>>(W1,   w1b,   2048 * 512);
    cvt_bf16_k<<<4096, 256, 0, stream>>>(W2,   w2b,   512 * 2048);
    cvt_bf16_k<<<64,   256, 0, stream>>>(Wt,   wtb,   32 * 512);

    // 1. embedding gather
    embed_gather_k<<<S_LEN, 256, 0, stream>>>(sent, embed, x, xb);

    // 2. QKV projection -> bf16 [2048,1536]
    gemm_wmma_k<false, false, true>
        <<<dim3(1536 / 64, S_LEN / 128), 256, 0, stream>>>(
            xb, wqkvb, bqkv, nullptr, qkvb, S_LEN, 3 * D_DIM, D_DIM);

    // 3. transpose V
    transpose_v_k<<<(H_NUM * S_LEN * DH) / 256, 256, 0, stream>>>(qkvb, vt);

    // 4. flash attention -> ob bf16
    flash_attn_k<<<dim3(S_LEN / 16, H_NUM), 32, 0, stream>>>(qkvb, vt, ob);

    // 5. output projection -> po f32
    gemm_wmma_k<false, true, false>
        <<<dim3(D_DIM / 64, S_LEN / 128), 256, 0, stream>>>(
            ob, wob, bo, po, nullptr, S_LEN, D_DIM, D_DIM);

    // 6. LN1(x + po) -> h1 f32 + h1b bf16
    ln_residual_k<<<S_LEN, 256, 0, stream>>>(x, po, ln1g, ln1b, h1, h1b);

    // 7. FFN up + ReLU -> a1 bf16
    gemm_wmma_k<true, false, true>
        <<<dim3(FF_DIM / 64, S_LEN / 128), 256, 0, stream>>>(
            h1b, w1b, b1, nullptr, a1, S_LEN, FF_DIM, D_DIM);

    // 8. FFN down -> f2 f32
    gemm_wmma_k<false, true, false>
        <<<dim3(D_DIM / 64, S_LEN / 128), 256, 0, stream>>>(
            a1, w2b, b2, f2, nullptr, S_LEN, D_DIM, FF_DIM);

    // 9. LN2(h1 + f2) -> h2b bf16
    ln_residual_k<<<S_LEN, 256, 0, stream>>>(h1, f2, ln2g, ln2b, nullptr, h2b);

    // 10. emission scores -> feats [2048,32] (OOB rows zero-filled by TDM)
    gemm_wmma_k<false, true, false>
        <<<dim3(1, S_LEN / 128), 256, 0, stream>>>(
            h2b, wtb, bt, feats, nullptr, S_LEN, L_TAG, D_DIM);

    // 11. Viterbi decode -> out[0]=score, out[1..2048]=path
    viterbi_k<<<1, 32, 0, stream>>>(feats, trans, bps, out);

    (void)in_sizes; (void)n_in; (void)out_size; (void)ws_size;
}
